// Net_53377853554780
// MI455X (gfx1250) — 8-bit matrix operand order repaired and checked by reading, compiled, not run on hardware
//
#include <hip/hip_runtime.h>
#include <hip/hip_bf16.h>

typedef int v8i __attribute__((ext_vector_type(8)));

struct SMem {
    unsigned bits[1808];
    unsigned a2[256 * 32];
    unsigned w3pk[512];
    unsigned w1m[4];
    unsigned long long w2m[8];
    signed char wfc1s[128];
    signed char wfc2s[8];
    signed char c3s[256 * 16];
};

__device__ __forceinline__ int isign(int v) {
    return min(1, max(-1, v));
}

__global__ __launch_bounds__(256) void binarynet_fused(
    const float* __restrict__ x,
    const float* __restrict__ w1,
    const float* __restrict__ w2,
    const float* __restrict__ w3,
    const float* __restrict__ wfc1,
    const float* __restrict__ wfc2,
    float* __restrict__ out)
{
    __shared__ SMem s;

    const int tid  = threadIdx.x;
    const int wave = tid >> 5;
    const int lane = tid & 31;

    if (tid < 4) {
        unsigned m = 0;
        #pragma unroll
        for (int k = 0; k < 9; ++k) m |= (unsigned)(w1[tid * 9 + k] > 0.0f) << k;
        s.w1m[tid] = m;
    }
    if (tid < 8) {
        unsigned long long m = 0;
        #pragma unroll
        for (int k = 0; k < 36; ++k)
            m |= (unsigned long long)(w2[tid * 36 + k] > 0.0f) << k;
        s.w2m[tid] = m;
    }
    for (int idx = tid; idx < 512; idx += 256) {
        int N = idx >> 5, Kd = idx & 31;
        unsigned d = 0;
        #pragma unroll
        for (int b = 0; b < 4; ++b) {
            int K = Kd * 4 + b;
            signed char v = (K < 72) ? (signed char)((w3[N * 72 + K] > 0.0f) ? 1 : -1) : (signed char)0;
            d |= (unsigned)(unsigned char)v << (8 * b);
        }
        s.w3pk[idx] = d;
    }
    if (tid < 32) {
        #pragma unroll
        for (int b = 0; b < 4; ++b) {
            int i = tid * 4 + b;
            s.wfc1s[i] = (signed char)((wfc1[i] > 0.0f) ? 1 : -1);
        }
    }
    if (tid == 0) {
        #pragma unroll
        for (int j = 0; j < 8; ++j)
            s.wfc2s[j] = (signed char)((wfc2[j] > 0.0f) ? 1 : -1);
    }

    const float* xc = x + (size_t)blockIdx.x * (256u * 225u);
    for (int k = 0; k < 225; ++k) {
        float v = xc[k * 256 + tid];
        unsigned m = __builtin_amdgcn_ballot_w32(v > 0.0f);
        if (lane == 0) s.bits[k * 8 + wave] = m;
    }
    __syncthreads();

    {
        const int off = tid * 225;
        const int dw = off >> 5;
        const int sh = off & 31;
        unsigned r[8];
        #pragma unroll
        for (int i = 0; i < 8; ++i) {
            unsigned lo = s.bits[dw + i];
            unsigned hi = s.bits[dw + i + 1];
            r[i] = sh ? ((lo >> sh) | (hi << (32 - sh))) : lo;
        }
        unsigned row[15];
        #pragma unroll
        for (int y = 0; y < 15; ++y) {
            int bp = y * 15, i = bp >> 5, s2 = bp & 31;
            unsigned lo = r[i], hi = r[i + 1 < 8 ? i + 1 : 7];
            row[y] = (s2 ? ((lo >> s2) | (hi << (32 - s2))) : lo) & 0x7FFFu;
        }

        unsigned c1r[4][7];
        #pragma unroll
        for (int oy = 0; oy < 7; ++oy) {
            unsigned ra0 = 0, ra1 = 0, ra2 = 0, ra3 = 0;
            #pragma unroll
            for (int ox = 0; ox < 7; ++ox) {
                unsigned p9 = ((row[2 * oy]     >> (2 * ox)) & 7u)
                            | (((row[2 * oy + 1] >> (2 * ox)) & 7u) << 3)
                            | (((row[2 * oy + 2] >> (2 * ox)) & 7u) << 6);
                ra0 |= (unsigned)((__popc((~(p9 ^ s.w1m[0])) & 0x1FFu) + 3) >> 3) << ox;
                ra1 |= (unsigned)((__popc((~(p9 ^ s.w1m[1])) & 0x1FFu) + 3) >> 3) << ox;
                ra2 |= (unsigned)((__popc((~(p9 ^ s.w1m[2])) & 0x1FFu) + 3) >> 3) << ox;
                ra3 |= (unsigned)((__popc((~(p9 ^ s.w1m[3])) & 0x1FFu) + 3) >> 3) << ox;
            }
            c1r[0][oy] = ra0; c1r[1][oy] = ra1; c1r[2][oy] = ra2; c1r[3][oy] = ra3;
        }

        unsigned long long pt[9];
        #pragma unroll
        for (int pos = 0; pos < 9; ++pos) {
            int oy = pos / 3, ox = pos % 3;
            unsigned long long p = 0;
            #pragma unroll
            for (int ic = 0; ic < 4; ++ic) {
                #pragma unroll
                for (int ky = 0; ky < 3; ++ky) {
                    unsigned rb = (c1r[ic][2 * oy + ky] >> (2 * ox)) & 7u;
                    p |= (unsigned long long)rb << (ic * 9 + ky * 3);
                }
            }
            pt[pos] = p;
        }

        unsigned a2loc[18];
        #pragma unroll
        for (int d = 0; d < 18; ++d) a2loc[d] = 0u;
        #pragma unroll
        for (int c = 0; c < 8; ++c) {
            #pragma unroll
            for (int pos = 0; pos < 9; ++pos) {
                int pc = __popcll((~(pt[pos] ^ s.w2m[c])) & 0xFFFFFFFFFull);
                int t = isign(pc - 18);
                int idx = c * 9 + pos;
                a2loc[idx >> 2] |= (unsigned)(unsigned char)(signed char)t << (8 * (idx & 3));
            }
        }
        uint4* dst = (uint4*)&s.a2[(unsigned)tid * 32u];
        dst[0] = make_uint4(a2loc[0],  a2loc[1],  a2loc[2],  a2loc[3]);
        dst[1] = make_uint4(a2loc[4],  a2loc[5],  a2loc[6],  a2loc[7]);
        dst[2] = make_uint4(a2loc[8],  a2loc[9],  a2loc[10], a2loc[11]);
        dst[3] = make_uint4(a2loc[12], a2loc[13], a2loc[14], a2loc[15]);
        dst[4] = make_uint4(a2loc[16], a2loc[17], 0u, 0u);
        dst[5] = make_uint4(0u, 0u, 0u, 0u);
        dst[6] = make_uint4(0u, 0u, 0u, 0u);
        dst[7] = make_uint4(0u, 0u, 0u, 0u);
    }
    __syncthreads();

    {
        const int half = lane >> 4;
        const int n    = lane & 15;
        v8i B0, B1;
        #pragma unroll
        for (int j = 0; j < 4; ++j) {
            int kd = half * 4 + j;
            B0[j] = (int)s.w3pk[n * 32 + kd];
            B1[j] = (int)s.w3pk[n * 32 + 16 + kd];
        }
        #pragma unroll
        for (int j = 4; j < 8; ++j) {
            int kd = 8 + half * 4 + (j - 4);
            B0[j] = (int)s.w3pk[n * 32 + kd];
            B1[j] = (int)s.w3pk[n * 32 + 16 + kd];
        }

        const int kmap[8] = {0, 1, 2, 3, 8, 9, 10, 11};
        #pragma unroll
        for (int tt = 0; tt < 2; ++tt) {
            const int tile = wave * 2 + tt;
            const unsigned* ap = &s.a2[(tile * 16 + n) * 32];
            v8i A0, A1;
            #pragma unroll
            for (int j = 0; j < 8; ++j) {
                int kd = kmap[j] + half * 4;
                A0[j] = (int)ap[kd];
                A1[j] = (int)ap[16 + kd];
            }
            v8i C = {0, 0, 0, 0, 0, 0, 0, 0};
            C = __builtin_amdgcn_wmma_i32_16x16x64_iu8(true, A0, true, B0, C, false, false);
            C = __builtin_amdgcn_wmma_i32_16x16x64_iu8(true, A1, true, B1, C, false, false);
            #pragma unroll
            for (int r = 0; r < 8; ++r) {
                int m = r + half * 8;
                s.c3s[(tile * 16 + m) * 16 + n] = (signed char)isign(C[r]);
            }
        }
    }
    __syncthreads();

    {
        const signed char* a = &s.c3s[tid * 16];
        int o = 0;
        #pragma unroll
        for (int j = 0; j < 8; ++j) {
            int h = 0;
            #pragma unroll
            for (int i = 0; i < 16; ++i) h += (int)a[i] * (int)s.wfc1s[j * 16 + i];
            o += isign(h) * (int)s.wfc2s[j];
        }
        out[(size_t)blockIdx.x * 256 + tid] = (float)o;
    }
}

extern "C" void kernel_launch(void* const* d_in, const int* in_sizes, int n_in,
                              void* d_out, int out_size, void* d_ws, size_t ws_size,
                              hipStream_t stream) {
    const float* x    = (const float*)d_in[0];
    const float* w1   = (const float*)d_in[1];
    const float* w2   = (const float*)d_in[2];
    const float* w3   = (const float*)d_in[3];
    const float* wfc1 = (const float*)d_in[4];
    const float* wfc2 = (const float*)d_in[5];
    float* outp = (float*)d_out;

    const int B = in_sizes[0] / 225;
    const int blocks = B / 256;
    binarynet_fused<<<blocks, 256, 0, stream>>>(x, w1, w2, w3, wfc1, wfc2, outp);
}
